// SurpriseGatedDeltaRecurrence_78847009620424
// MI455X (gfx1250) — compile-verified, matrix operand order checked by reading, not run on hardware
//
#include <hip/hip_runtime.h>
#include <hip/hip_bf16.h>
#include <math.h>

// Problem constants
#define BB   8
#define SS   2048
#define DD   1024
#define DSS  128
#define MM   (BB * SS)     // 16384 tokens

typedef __attribute__((ext_vector_type(16))) __bf16 v16bf;
typedef __attribute__((ext_vector_type(8)))  float  v8f;

// ---------------------------------------------------------------------------
// float -> bf16: native gfx1250 conversion (v_cvt_pk_bf16_f32)
// ---------------------------------------------------------------------------
__device__ __forceinline__ __bf16 f2bf(float f) { return (__bf16)f; }

// Raw (still-f32) 16x32 fragment data for one lane: floats {0..7} and {16..23}
// relative to rowPtr (which already includes the +half*8 lane offset).
struct Raw {
    float4 r0, r1, r2, r3;
};
__device__ __forceinline__ Raw load_raw(const float* rowPtr) {
    const float4* p = (const float4*)rowPtr;   // 32B aligned
    Raw r;
    r.r0 = p[0];   // floats 0..3
    r.r1 = p[1];   // floats 4..7
    r.r2 = p[4];   // floats 16..19
    r.r3 = p[5];   // floats 20..23
    return r;
}
__device__ __forceinline__ v16bf cvt_frag(const Raw& r) {
    v16bf f;
    f[0]  = f2bf(r.r0.x); f[1]  = f2bf(r.r0.y); f[2]  = f2bf(r.r0.z); f[3]  = f2bf(r.r0.w);
    f[4]  = f2bf(r.r1.x); f[5]  = f2bf(r.r1.y); f[6]  = f2bf(r.r1.z); f[7]  = f2bf(r.r1.w);
    f[8]  = f2bf(r.r2.x); f[9]  = f2bf(r.r2.y); f[10] = f2bf(r.r2.z); f[11] = f2bf(r.r2.w);
    f[12] = f2bf(r.r3.x); f[13] = f2bf(r.r3.y); f[14] = f2bf(r.r3.z); f[15] = f2bf(r.r3.w);
    return f;
}

// ---------------------------------------------------------------------------
// Async LDS load (gfx1250): one dword per lane, tracked by ASYNCcnt.
// ---------------------------------------------------------------------------
__device__ __forceinline__ void async_ld_b32(void* lds_ptr, const void* gptr) {
    unsigned lds = (unsigned)(uintptr_t)lds_ptr;   // flat low 32 bits = LDS addr
    asm volatile("global_load_async_to_lds_b32 %0, %1, off"
                 :: "v"(lds), "v"(gptr)
                 : "memory");
}
__device__ __forceinline__ void wait_async0() {
    asm volatile("s_wait_asynccnt 0" ::: "memory");
}

// ---------------------------------------------------------------------------
// Kernel 1: fused projection GEMM.  proj[m][n] (ldc=512):
// n in [0,128)=k_raw, [128,256)=v, [256,384)=q_raw, [384,512)=r_raw.
// One 16x64 tile per wave.  Per K-step: issue ALL 20 b128 loads first
// (one VMEM clause), then convert, then 4 back-to-back WMMAs.
// ---------------------------------------------------------------------------
__global__ __launch_bounds__(256) void proj_gemm(
        const float* __restrict__ X,
        const float* __restrict__ Wk, const float* __restrict__ Wv,
        const float* __restrict__ Wq, const float* __restrict__ Wa,
        float* __restrict__ Cout) {
    const int K = DD, LDC = 4 * DSS;          // 1024, 512
    const int lane = threadIdx.x & 31;
    const int tile = blockIdx.x * 8 + (threadIdx.x >> 5);
    const int m0 = (tile >> 3) * 16;          // 8 n-supertiles of width 64
    const int n0 = (tile & 7) * 64;

    const float* W = (n0 < 128) ? Wk : (n0 < 256) ? Wv : (n0 < 384) ? Wq : Wa;
    const int nloc = (n0 & 127) + (lane & 15);
    const int half = (lane >> 4);

    const float* arow = X + (size_t)(m0 + (lane & 15)) * K + half * 8;
    const float* brow = W + (size_t)nloc * K + half * 8;     // + tt*16 rows

    v8f acc0 = {0.f,0.f,0.f,0.f,0.f,0.f,0.f,0.f};
    v8f acc1 = acc0, acc2 = acc0, acc3 = acc0;

    for (int kb = 0; kb < K; kb += 32) {
        // ---- load phase: 20 independent global_load_b128 ----
        Raw ra  = load_raw(arow + kb);
        Raw rb0 = load_raw(brow + kb);
        Raw rb1 = load_raw(brow + (size_t)16 * K + kb);
        Raw rb2 = load_raw(brow + (size_t)32 * K + kb);
        Raw rb3 = load_raw(brow + (size_t)48 * K + kb);
        // ---- convert phase ----
        v16bf a  = cvt_frag(ra);
        v16bf b0 = cvt_frag(rb0);
        v16bf b1 = cvt_frag(rb1);
        v16bf b2 = cvt_frag(rb2);
        v16bf b3 = cvt_frag(rb3);
        // ---- matrix phase ----
        acc0 = __builtin_amdgcn_wmma_f32_16x16x32_bf16(false, a, false, b0, (short)0, acc0, false, false);
        acc1 = __builtin_amdgcn_wmma_f32_16x16x32_bf16(false, a, false, b1, (short)0, acc1, false, false);
        acc2 = __builtin_amdgcn_wmma_f32_16x16x32_bf16(false, a, false, b2, (short)0, acc2, false, false);
        acc3 = __builtin_amdgcn_wmma_f32_16x16x32_bf16(false, a, false, b3, (short)0, acc3, false, false);
    }
    // C layout: lanes 0-15 -> M = m0+reg, lanes 16-31 -> M = m0+8+reg
    const int mrow = m0 + (half ? 8 : 0);
    float* cp = Cout + (size_t)mrow * LDC + n0 + (lane & 15);
#pragma unroll
    for (int i = 0; i < 8; ++i) {
        cp[(size_t)i * LDC +  0] = acc0[i];
        cp[(size_t)i * LDC + 16] = acc1[i];
        cp[(size_t)i * LDC + 32] = acc2[i];
        cp[(size_t)i * LDC + 48] = acc3[i];
    }
}

// ---------------------------------------------------------------------------
// Kernel 2: per-token post-processing (l2norm k/q, alpha). One wave per token.
// ---------------------------------------------------------------------------
__global__ __launch_bounds__(256) void postproc(
        const float* __restrict__ proj,
        const float* __restrict__ bias,      // W_alpha_b [128]
        const float* __restrict__ lam,       // lambda_param [128]
        float* __restrict__ kn, float* __restrict__ qn,
        float* __restrict__ al) {
    const int lane = threadIdx.x & 31;
    const int row  = blockIdx.x * 8 + (threadIdx.x >> 5);
    const float* pr = proj + (size_t)row * 512;

    float kv[4], qv[4], sk = 0.f, sq = 0.f;
#pragma unroll
    for (int i = 0; i < 4; ++i) {
        const int j = lane + i * 32;
        kv[i] = pr[j];        sk += kv[i] * kv[i];
        qv[i] = pr[256 + j];  sq += qv[i] * qv[i];
    }
#pragma unroll
    for (int off = 16; off; off >>= 1) {
        sk += __shfl_xor(sk, off, 32);
        sq += __shfl_xor(sq, off, 32);
    }
    const float ik = 1.f / fmaxf(sqrtf(sk), 1e-12f);
    const float iq = 1.f / fmaxf(sqrtf(sq), 1e-12f);
#pragma unroll
    for (int i = 0; i < 4; ++i) {
        const int j = lane + i * 32;
        kn[(size_t)row * 128 + j] = kv[i] * ik;
        qn[(size_t)row * 128 + j] = qv[i] * iq;
        const float rr  = pr[384 + j] + bias[j];
        const float sig = 1.f / (1.f + __expf(-rr));
        const float sl  = 1.f / (1.f + __expf(-lam[j]));
        const float la  = __logf(sl + 1e-8f);
        al[(size_t)row * 128 + j] = __expf(8.0f * sig * la);
    }
}

// ---------------------------------------------------------------------------
// Kernel 3: sequential surprise-gated delta scan. One workgroup per batch,
// 512 threads, H (128x128) in registers: thread (v = tid&127, quarter =
// tid>>7) owns H[k0..k0+31][v].  Step vectors (q/k/v/alpha) are DOUBLE-
// BUFFERED in LDS via gfx1250 async loads: issue t+1 at the top of step t,
// s_wait_asynccnt 0 + barrier at the bottom.
// ---------------------------------------------------------------------------
__global__ __launch_bounds__(512) void scan_kernel(
        const float* __restrict__ proj,   // v at column offset 128, ld 512
        const float* __restrict__ kn,
        const float* __restrict__ qn,
        const float* __restrict__ al,
        const float* __restrict__ norm_w,
        float* __restrict__ yn) {
    __shared__ float vb[2][4][128];        // [buf][q,k,v,a][128]
    __shared__ float cs[128], nw[128];
    __shared__ float part[512], part2[512], yP[512];
    __shared__ float errW[4], msW[4];

    const int b    = blockIdx.x;
    const int tid  = threadIdx.x;
    const int v    = tid & 127;
    const int qr   = tid >> 7;            // quarter 0..3
    const int k0   = qr * 32;
    const int lane = tid & 31;
    const int wave = tid >> 5;

    float h[32];
#pragma unroll
    for (int r = 0; r < 32; ++r) h[r] = 0.f;

    if (tid < 128) nw[tid] = norm_w[tid];

    const int sel = qr;                    // which vector this thread loads
    const int j   = v;

    // source pointer for (b, t, sel, j)
    auto src_of = [&](int t) -> const float* {
        const size_t bs = (size_t)b * SS + t;
        if      (sel == 0) return &qn[bs * 128 + j];
        else if (sel == 1) return &kn[bs * 128 + j];
        else if (sel == 2) return &proj[bs * 512 + 128 + j];
        else               return &al[bs * 128 + j];
    };

    // prologue: async-load step 0 into buffer 0
    async_ld_b32(&vb[0][sel][j], src_of(0));
    wait_async0();
    __syncthreads();

    for (int t = 0; t < SS; ++t) {
        const int p = t & 1;
        const size_t base = (size_t)b * SS + t;

        // kick off next step's vectors into the other buffer (async DMA)
        if (t + 1 < SS) async_ld_b32(&vb[p ^ 1][sel][j], src_of(t + 1));
        if (t + 2 < SS) __builtin_prefetch(src_of(t + 2), 0, 1);

        const float* qs  = vb[p][0];
        const float* ks_ = vb[p][1];
        const float* vs  = vb[p][2];
        const float* as_ = vb[p][3];

        // per-row gate coefficient c_k = (1 - a_k) * k_k
        if (tid < 128) cs[tid] = (1.f - as_[tid]) * ks_[tid];

        // -- pass 1: pred_v = q.H, kproj_v = k.H (pre-update H) -------------
        float pa = 0.f, pk = 0.f;
#pragma unroll
        for (int r = 0; r < 32; ++r) {
            const float hv = h[r];
            pa += qs[k0 + r] * hv;
            pk += ks_[k0 + r] * hv;
        }
        part[tid]  = pa;
        part2[tid] = pk;
        __syncthreads();

        const float pred  = part[v]  + part[128 + v]  + part[256 + v]  + part[384 + v];
        const float kproj = part2[v] + part2[128 + v] + part2[256 + v] + part2[384 + v];

        if (qr == 0) {                       // err = sum_v (v_v - pred_v)^2
            float d = vs[v] - pred;
            float e = d * d;
#pragma unroll
            for (int off = 16; off; off >>= 1) e += __shfl_xor(e, off, 32);
            if (lane == 0) errW[wave] = e;   // waves 0..3
        }
        __syncthreads();

        const float err = errW[0] + errW[1] + errW[2] + errW[3];
        const float sur = 1.f / (1.f + __expf(-err * (1.f / (1.0f + 1e-6f))));
        const float u   = sur * (vs[v] - kproj);   // per-column update scalar

        // -- pass 2: H = a*H + c*u  fused with  y = q . H_new ---------------
        float ya = 0.f;
#pragma unroll
        for (int r = 0; r < 32; ++r) {
            const float hnew = as_[k0 + r] * h[r] + cs[k0 + r] * u;
            h[r] = hnew;
            ya += qs[k0 + r] * hnew;
        }
        yP[tid] = ya;
        __syncthreads();

        float yv = 0.f;
        if (qr == 0) {
            yv = yP[v] + yP[128 + v] + yP[256 + v] + yP[384 + v];
            float e = yv * yv;
#pragma unroll
            for (int off = 16; off; off >>= 1) e += __shfl_xor(e, off, 32);
            if (lane == 0) msW[wave] = e;
        }
        __syncthreads();

        if (qr == 0) {                       // fused RMS-norm + scale
            const float ms = (msW[0] + msW[1] + msW[2] + msW[3]) * (1.f / 128.f);
            const float sc = rsqrtf(ms + 1e-6f);
            yn[base * 128 + v] = yv * sc * nw[v];
        }

        // ensure next step's async data has landed + all LDS readers done
        wait_async0();
        __syncthreads();
    }
}

// ---------------------------------------------------------------------------
// Kernel 4: output GEMM  out[16384x1024] = yn[16384x128] . W_o^T.
// One 16x64 tile per wave, K=128.  Same load/convert/matrix phasing.
// ---------------------------------------------------------------------------
__global__ __launch_bounds__(256) void out_gemm(
        const float* __restrict__ Y,       // [M,128]
        const float* __restrict__ Wo,      // [1024,128]
        float* __restrict__ Out) {         // [M,1024]
    const int K = DSS, LDC = DD;           // 128, 1024
    const int lane = threadIdx.x & 31;
    const int tile = blockIdx.x * 8 + (threadIdx.x >> 5);
    const int m0 = (tile >> 4) * 16;       // 16 n-supertiles of width 64
    const int n0 = (tile & 15) * 64;
    const int half = (lane >> 4);

    const float* arow = Y  + (size_t)(m0 + (lane & 15)) * K + half * 8;
    const float* brow = Wo + (size_t)(n0 + (lane & 15)) * K + half * 8;

    v8f acc0 = {0.f,0.f,0.f,0.f,0.f,0.f,0.f,0.f};
    v8f acc1 = acc0, acc2 = acc0, acc3 = acc0;
#pragma unroll
    for (int kb = 0; kb < K; kb += 32) {
        Raw ra  = load_raw(arow + kb);
        Raw rb0 = load_raw(brow + kb);
        Raw rb1 = load_raw(brow + (size_t)16 * K + kb);
        Raw rb2 = load_raw(brow + (size_t)32 * K + kb);
        Raw rb3 = load_raw(brow + (size_t)48 * K + kb);
        v16bf a  = cvt_frag(ra);
        v16bf b0 = cvt_frag(rb0);
        v16bf b1 = cvt_frag(rb1);
        v16bf b2 = cvt_frag(rb2);
        v16bf b3 = cvt_frag(rb3);
        acc0 = __builtin_amdgcn_wmma_f32_16x16x32_bf16(false, a, false, b0, (short)0, acc0, false, false);
        acc1 = __builtin_amdgcn_wmma_f32_16x16x32_bf16(false, a, false, b1, (short)0, acc1, false, false);
        acc2 = __builtin_amdgcn_wmma_f32_16x16x32_bf16(false, a, false, b2, (short)0, acc2, false, false);
        acc3 = __builtin_amdgcn_wmma_f32_16x16x32_bf16(false, a, false, b3, (short)0, acc3, false, false);
    }
    const int mrow = m0 + (half ? 8 : 0);
    float* cp = Out + (size_t)mrow * LDC + n0 + (lane & 15);
#pragma unroll
    for (int i = 0; i < 8; ++i) {
        cp[(size_t)i * LDC +  0] = acc0[i];
        cp[(size_t)i * LDC + 16] = acc1[i];
        cp[(size_t)i * LDC + 32] = acc2[i];
        cp[(size_t)i * LDC + 48] = acc3[i];
    }
}

// ---------------------------------------------------------------------------
extern "C" void kernel_launch(void* const* d_in, const int* in_sizes, int n_in,
                              void* d_out, int out_size, void* d_ws, size_t ws_size,
                              hipStream_t stream) {
    const float* x   = (const float*)d_in[0];
    const float* Wk  = (const float*)d_in[1];
    const float* Wv  = (const float*)d_in[2];
    const float* Wq  = (const float*)d_in[3];
    const float* Waw = (const float*)d_in[4];
    const float* Wab = (const float*)d_in[5];
    const float* lam = (const float*)d_in[6];
    const float* Wo  = (const float*)d_in[7];
    const float* nw  = (const float*)d_in[8];
    float* out = (float*)d_out;

    float* proj = (float*)d_ws;                            // 16384*512
    float* knb  = proj + (size_t)MM * 512;                 // 16384*128
    float* qnb  = knb  + (size_t)MM * 128;
    float* alb  = qnb  + (size_t)MM * 128;
    float* ynb  = alb  + (size_t)MM * 128;

    // 1) projections: (16384/16) m-tiles * 8 n-supertiles = 8192 waves / 8
    proj_gemm<<<1024, 256, 0, stream>>>(x, Wk, Wv, Wq, Waw, proj);
    // 2) l2norm + alpha: 16384 tokens, 8 per block (wave-per-token)
    postproc<<<2048, 256, 0, stream>>>(proj, Wab, lam, knb, qnb, alb);
    // 3) sequential scan: one WG per batch
    scan_kernel<<<BB, 512, 0, stream>>>(proj, knb, qnb, alb, nw, ynb);
    // 4) output GEMM: (16384/16) m-tiles * 16 n-supertiles = 16384 waves / 8
    out_gemm<<<2048, 256, 0, stream>>>(ynb, Wo, out);
}
